// FlowerBrainPG_70360154243403
// MI455X (gfx1250) — compile-verified, matrix operand order checked by reading, not run on hardware
//
#include <hip/hip_runtime.h>
#include <hip/hip_bf16.h>
#include <stdint.h>

// ---------------------------------------------------------------------------
// MI455X (gfx1250) forward pass for the BitNet-style transformer reference.
// GEMMs: TDM (tensor_load_to_lds) double-buffered LDS staging feeding
// v_wmma_f32_16x16x32_bf16; attention QK^T / PV also on WMMA.
// Workspace requirement: ~132 MB.
// ---------------------------------------------------------------------------

typedef __bf16 bf16_t;
typedef __attribute__((ext_vector_type(16))) __bf16 v16bf;
typedef __attribute__((ext_vector_type(8)))  float  v8f;
typedef unsigned int v4u __attribute__((ext_vector_type(4)));
typedef int          v8i __attribute__((ext_vector_type(8)));
typedef int          v4i __attribute__((ext_vector_type(4)));

#if __has_builtin(__builtin_amdgcn_tensor_load_to_lds) && \
    __has_builtin(__builtin_amdgcn_s_wait_tensorcnt)
#define FB_USE_TDM 1
#else
#define FB_USE_TDM 0
#endif

namespace {

constexpr int kV = 8192, kD = 1024, kH = 16, kHD = 64, kKVD = 512, kHID = 4096;
constexpr int kNL = 11, kBigB = 3072, kBigD = 112, kBigP = 128;
constexpr int kB = 2, kT = 512, kM = kB * kT, kZH = kB * kH;
constexpr int kKc = 128;            // K chunk staged per TDM transfer
constexpr float kEps = 1.1920929e-7f;
constexpr float kCap = 30.0f;

// ---------------- WMMA fragment helpers (wave32 layouts, ISA 7.12.2) -------

// A-fragment / row-major-K B-fragment: 16 rows x 32 K, bf16.
// lane<16 : row=lane,   K-runs [0..7] and [16..23]
// lane>=16: row=lane-16, K-runs [8..15] and [24..31]  -> two 16B loads.
__device__ inline v16bf frag_rowK(const bf16_t* base, int ld) {
  int lane = threadIdx.x & 31;
  int r    = lane & 15;
  int kh   = (lane & 16) ? 8 : 0;
  const bf16_t* p = base + (size_t)r * ld + kh;
  union { uint4 u[2]; v16bf v; } u;
  u.u[0] = *(const uint4*)(p);
  u.u[1] = *(const uint4*)(p + 16);
  return u.v;
}

// B-fragment for row-major K x N matrix (32 K x 16 N): lane holds a column.
__device__ inline v16bf frag_colN(const bf16_t* base, int ld) {
  int lane = threadIdx.x & 31;
  int c    = lane & 15;
  int kh   = (lane & 16) ? 8 : 0;
  v16bf b;
#pragma unroll
  for (int i = 0; i < 8; ++i) {
    b[i]     = base[(size_t)(kh + i)      * ld + c];
    b[8 + i] = base[(size_t)(kh + 16 + i) * ld + c];
  }
  return b;
}

// C/D store: VGPR v -> (M = v or v+8 by lane group), N = lane&15.
__device__ inline void store_acc(float* C, int ldc, int m0, int n0, const v8f& c) {
  int lane = threadIdx.x & 31;
  int cr = (lane & 16) ? 8 : 0;
  int cc = lane & 15;
#pragma unroll
  for (int v = 0; v < 8; ++v)
    C[(size_t)(m0 + cr + v) * ldc + n0 + cc] = c[v];
}

// ---------------- Tensor Data Mover 2-D descriptor (ISA ch8, D# groups) ----

#if FB_USE_TDM
// Load a [tile_rows x tile_k] bf16 tile (row stride = ld elements) from global
// into LDS at lds_addr. Normal mode, 2-D (groups 2/3 zero), data_size = 2B.
__device__ inline void tdm_load_2d(uint32_t lds_addr, const bf16_t* gptr,
                                   int tile_k, int tile_rows,
                                   int tensor_k, int tensor_rows, int ld) {
  uint64_t ga = (uint64_t)(uintptr_t)gptr;
  v4u g0;
  g0[0] = 1u;                                              // count=1, user mode
  g0[1] = lds_addr;                                        // LDS byte address
  g0[2] = (uint32_t)ga;                                    // global addr[31:0]
  g0[3] = (uint32_t)((ga >> 32) & 0x01FFFFFFu) | (2u << 30); // addr[56:32]|type=2
  v8i g1;
  g1[0] = 0x00010000;                          // data_size=1(2B); no mcast/pad
  g1[1] = (int)((uint32_t)tensor_k << 16);                 // tensor_dim0 lo16
  g1[2] = (int)(((uint32_t)tensor_k >> 16) |
                ((uint32_t)tensor_rows << 16));            // dim0 hi | dim1 lo
  g1[3] = (int)(((uint32_t)tensor_rows >> 16) |
                ((uint32_t)tile_k << 16));                 // dim1 hi | tile_dim0
  g1[4] = (int)((uint32_t)tile_rows & 0xFFFFu);            // tile_dim1 (dim2=0)
  g1[5] = (int)(uint32_t)ld;                               // dim0_stride[31:0]
  g1[6] = 0;                                               // stride hi / d1stride
  g1[7] = 0;
  v4i z4 = {};
#if __clang_major__ >= 23
  v8i z8 = {};
  __builtin_amdgcn_tensor_load_to_lds(g0, g1, z4, z4, z8, 0);
#else
  __builtin_amdgcn_tensor_load_to_lds(g0, g1, z4, z4, 0);
#endif
}
#endif  // FB_USE_TDM

// ---------------- main GEMM: C[M,N] = A[M,K] * B[N,K]^T --------------------
// Block = 4 waves stacked on M (64M x 64N block tile). A and B panels are
// staged into LDS by the TDM (double buffered, Kc=128 chunks); WMMA fragments
// are read with ds_load_b128. Requires M%64==0, N%64==0, K%128==0.
__global__ void __launch_bounds__(128)
gemm_rr_kernel(const bf16_t* __restrict__ A, int lda,
               const bf16_t* __restrict__ Bm, int ldb,
               float* __restrict__ C, int ldc, int M, int N, int K) {
  __shared__ bf16_t As[2][64 * kKc];   // 16 KB x2
  __shared__ bf16_t Bs[2][64 * kKc];   // 16 KB x2
  int wave = threadIdx.x >> 5;
  int m0 = blockIdx.y * 64;
  int n0 = blockIdx.x * 64;
  int nstages = K / kKc;
  v8f acc[4] = {};

#if FB_USE_TDM
  if (wave == 0) {
    tdm_load_2d((uint32_t)(uintptr_t)(void*)As[0], A + (size_t)m0 * lda,
                kKc, 64, K, M, lda);
    tdm_load_2d((uint32_t)(uintptr_t)(void*)Bs[0], Bm + (size_t)n0 * ldb,
                kKc, 64, K, N, ldb);
  }
#endif
  for (int s = 0; s < nstages; ++s) {
    int buf = s & 1;
#if FB_USE_TDM
    if (wave == 0) {
      if (s + 1 < nstages) {   // prefetch next stage into back buffer
        tdm_load_2d((uint32_t)(uintptr_t)(void*)As[buf ^ 1],
                    A + (size_t)m0 * lda + (s + 1) * kKc, kKc, 64, K, M, lda);
        tdm_load_2d((uint32_t)(uintptr_t)(void*)Bs[buf ^ 1],
                    Bm + (size_t)n0 * ldb + (s + 1) * kKc, kKc, 64, K, N, ldb);
        __builtin_amdgcn_s_wait_tensorcnt(2);  // current stage's 2 loads done
      } else {
        __builtin_amdgcn_s_wait_tensorcnt(0);
      }
    }
    __syncthreads();
#else
    // Fallback: cooperative vectorized copy (8 bf16 per uint4).
    {
      const bf16_t* gA = A + (size_t)m0 * lda + s * kKc;
      const bf16_t* gB = Bm + (size_t)n0 * ldb + s * kKc;
      for (int i = threadIdx.x; i < 64 * (kKc / 8); i += 128) {
        int r = i / (kKc / 8), c8 = (i % (kKc / 8)) * 8;
        *(uint4*)&As[buf][r * kKc + c8] = *(const uint4*)&gA[(size_t)r * lda + c8];
        *(uint4*)&Bs[buf][r * kKc + c8] = *(const uint4*)&gB[(size_t)r * ldb + c8];
      }
    }
    __syncthreads();
#endif
    const bf16_t* Apan = As[buf] + wave * 16 * kKc;   // this wave's 16 M-rows
#pragma unroll
    for (int kk = 0; kk < kKc; kk += 32) {
      v16bf a = frag_rowK(Apan + kk, kKc);
#pragma unroll
      for (int j = 0; j < 4; ++j) {
        v16bf b = frag_rowK(Bs[buf] + (size_t)(j * 16) * kKc + kk, kKc);
        acc[j] = __builtin_amdgcn_wmma_f32_16x16x32_bf16(
            false, a, false, b, (short)0, acc[j], false, false);
      }
    }
    __syncthreads();
  }
  int mw = m0 + wave * 16;
#pragma unroll
  for (int j = 0; j < 4; ++j) store_acc(C, ldc, mw, n0 + j * 16, acc[j]);
}

// ---------------- attention WMMA kernels -----------------------------------

// scores[z,i,j] = q_i . k_j per (batch,head); tiles above diagonal skipped.
__global__ void attn_scores_kernel(const bf16_t* __restrict__ qb,
                                   const bf16_t* __restrict__ kb,
                                   float* __restrict__ scores) {
  int wave = threadIdx.x >> 5;
  int z = blockIdx.z, b = z / kH, h = z % kH, kvh = h >> 1;
  int m0 = blockIdx.y * 16;
  int n0 = (blockIdx.x * 4 + wave) * 64;
  if (n0 > m0 + 15) return;
  const bf16_t* A  = qb + (size_t)b * kT * kD   + (size_t)h   * kHD;
  const bf16_t* Bm = kb + (size_t)b * kT * kKVD + (size_t)kvh * kHD;
  float* C = scores + (size_t)z * kT * kT;
  v8f acc[4] = {};
#pragma unroll
  for (int k0 = 0; k0 < kHD; k0 += 32) {
    v16bf a = frag_rowK(A + (size_t)m0 * kD + k0, kD);
#pragma unroll
    for (int j = 0; j < 4; ++j) {
      v16bf bfr = frag_rowK(Bm + (size_t)(n0 + j * 16) * kKVD + k0, kKVD);
      acc[j] = __builtin_amdgcn_wmma_f32_16x16x32_bf16(
          false, a, false, bfr, (short)0, acc[j], false, false);
    }
  }
#pragma unroll
  for (int j = 0; j < 4; ++j) store_acc(C, kT, m0, n0 + j * 16, acc[j]);
}

// o[z,i,:64] = probs[z,i,:] * v[:,kvh*64..+64]; K loop stops at diagonal.
__global__ void attn_o_kernel(const bf16_t* __restrict__ probs,
                              const bf16_t* __restrict__ vb,
                              float* __restrict__ o32) {
  int z = blockIdx.z, b = z / kH, h = z % kH, kvh = h >> 1;
  int m0 = blockIdx.y * 16;
  const bf16_t* A  = probs + (size_t)z * kT * kT;
  const bf16_t* Bm = vb + (size_t)b * kT * kKVD + (size_t)kvh * kHD;
  float* C = o32 + (size_t)b * kT * kD + (size_t)h * kHD;
  v8f acc[4] = {};
  for (int k0 = 0; k0 < m0 + 16; k0 += 32) {   // probs zero above diagonal
    v16bf a = frag_rowK(A + (size_t)m0 * kT + k0, kT);
#pragma unroll
    for (int j = 0; j < 4; ++j) {
      v16bf bfr = frag_colN(Bm + (size_t)k0 * kKVD + j * 16, kKVD);
      acc[j] = __builtin_amdgcn_wmma_f32_16x16x32_bf16(
          false, a, false, bfr, (short)0, acc[j], false, false);
    }
  }
#pragma unroll
  for (int j = 0; j < 4; ++j) store_acc(C, kD, m0, j * 16, acc[j]);
}

// ---------------- quantization / normalization -----------------------------

__global__ void absmean_partial(const float* __restrict__ w,
                                float* __restrict__ partials, long long n) {
  __shared__ float red[256];
  float s = 0.f;
  for (long long i = (long long)blockIdx.x * 256 + threadIdx.x; i < n;
       i += 256LL * 256LL)
    s += fabsf(w[i]);
  red[threadIdx.x] = s;
  __syncthreads();
  for (int st = 128; st > 0; st >>= 1) {
    if ((int)threadIdx.x < st) red[threadIdx.x] += red[threadIdx.x + st];
    __syncthreads();
  }
  if (threadIdx.x == 0) partials[blockIdx.x] = red[0];
}

__global__ void absmean_final(const float* __restrict__ partials,
                              float* __restrict__ thr, float invn) {
  __shared__ float red[256];
  red[threadIdx.x] = partials[threadIdx.x];
  __syncthreads();
  for (int st = 128; st > 0; st >>= 1) {
    if ((int)threadIdx.x < st) red[threadIdx.x] += red[threadIdx.x + st];
    __syncthreads();
  }
  if (threadIdx.x == 0) thr[0] = red[0] * invn;
}

// W[N,K] fp32 -> ternary bf16 [N,Kpad] (zero padding in K).
__global__ void ternary_bf16_kernel(const float* __restrict__ w,
                                    const float* __restrict__ thr,
                                    bf16_t* __restrict__ out,
                                    int N, int K, int Kpad) {
  long long idx = (long long)blockIdx.x * 256 + threadIdx.x;
  long long total = (long long)N * Kpad;
  if (idx >= total) return;
  int n = (int)(idx / Kpad), k = (int)(idx % Kpad);
  float t = thr[0];
  float q = 0.f;
  if (k < K) {
    float v = w[(size_t)n * K + k];
    q = (fabsf(v) > t) ? (v > 0.f ? 1.f : -1.f) : 0.f;
  }
  out[idx] = (bf16_t)q;
}

// rmsnorm row and emit bf16 (K -> Kpad zero-padded).
__global__ void rmsnorm_bf16_kernel(const float* __restrict__ x,
                                    const float* __restrict__ g,
                                    bf16_t* __restrict__ out, int Dd, int Dpad) {
  __shared__ float red[256];
  int row = blockIdx.x, tid = threadIdx.x;
  const float* xr = x + (size_t)row * Dd;
  float s = 0.f;
  for (int d = tid; d < Dd; d += 256) { float v = xr[d]; s += v * v; }
  red[tid] = s;
  __syncthreads();
  for (int st = 128; st > 0; st >>= 1) {
    if (tid < st) red[tid] += red[tid + st];
    __syncthreads();
  }
  float r = rsqrtf(red[0] / (float)Dd + kEps);
  bf16_t* orow = out + (size_t)row * Dpad;
  for (int d = tid; d < Dpad; d += 256)
    orow[d] = (d < Dd) ? (bf16_t)(xr[d] * r * g[d]) : (bf16_t)0.f;
}

__global__ void rmsnorm_f32_kernel(const float* __restrict__ x,
                                   const float* __restrict__ g,
                                   float* __restrict__ out, int Dd) {
  __shared__ float red[256];
  int row = blockIdx.x, tid = threadIdx.x;
  const float* xr = x + (size_t)row * Dd;
  float s = 0.f;
  for (int d = tid; d < Dd; d += 256) { float v = xr[d]; s += v * v; }
  red[tid] = s;
  __syncthreads();
  for (int st = 128; st > 0; st >>= 1) {
    if (tid < st) red[tid] += red[tid + st];
    __syncthreads();
  }
  float r = rsqrtf(red[0] / (float)Dd + kEps);
  float* orow = out + (size_t)row * Dd;
  for (int d = tid; d < Dd; d += 256) orow[d] = xr[d] * r * g[d];
}

__global__ void f32_to_bf16_kernel(const float* __restrict__ src,
                                   bf16_t* __restrict__ dst, long long n) {
  long long i = (long long)blockIdx.x * 256 + threadIdx.x;
  if (i < n) dst[i] = (bf16_t)src[i];
}

// ---------------- elementwise / small kernels ------------------------------

__global__ void embed_kernel(const long long* __restrict__ tok,
                             const float* __restrict__ tok_emb,
                             const float* __restrict__ big_emb,
                             float* __restrict__ x, float* __restrict__ bg) {
  int m = blockIdx.x, tid = threadIdx.x;
  int t = m % kT;
  long long cur  = tok[m];
  long long prev = (t == 0) ? 0 : tok[m - 1];
  long long hsh  = ((prev * 16777619LL) ^ cur) % kBigB;  // FNV-1a style
  const float* te = tok_emb + (size_t)cur * kD;
  float* xr = x + (size_t)m * kD;
  for (int d = tid; d < kD; d += 256) xr[d] = te[d];
  const float* be = big_emb + (size_t)hsh * kBigD;
  float* br = bg + (size_t)m * kBigD;
  for (int d = tid; d < kBigD; d += 256) br[d] = be[d];
}

__global__ void add2_kernel(float* __restrict__ out, const float* __restrict__ a,
                            const float* __restrict__ b, long long n) {
  long long i = (long long)blockIdx.x * 256 + threadIdx.x;
  if (i < n) out[i] = a[i] + b[i];
}

__global__ void add3_kernel(float* __restrict__ x, const float* __restrict__ a,
                            const float* __restrict__ b, long long n) {
  long long i = (long long)blockIdx.x * 256 + threadIdx.x;
  if (i < n) x[i] = x[i] + a[i] + b[i];
}

// causal softmax; scale (qk_gain/sqrt(HD)) read from device scalar; writes
// bf16 probs with zeros above the diagonal.
__global__ void softmax_kernel(const float* __restrict__ scores,
                               bf16_t* __restrict__ probs,
                               const float* __restrict__ gain) {
  int i = blockIdx.x, z = blockIdx.y, lane = threadIdx.x;
  float scale = gain[0] * rsqrtf((float)kHD);
  const float* row = scores + ((size_t)z * kT + i) * kT;
  bf16_t* prow = probs + ((size_t)z * kT + i) * kT;
  float mx = -__builtin_inff();
  for (int j = lane; j <= i; j += 32) mx = fmaxf(mx, row[j] * scale);
  for (int off = 16; off; off >>= 1) mx = fmaxf(mx, __shfl_xor(mx, off, 32));
  float sum = 0.f;
  for (int j = lane; j <= i; j += 32) sum += __expf(row[j] * scale - mx);
  for (int off = 16; off; off >>= 1) sum += __shfl_xor(sum, off, 32);
  float inv = 1.f / sum;
  for (int j = lane; j < kT; j += 32) {
    float p = (j <= i) ? __expf(row[j] * scale - mx) * inv : 0.f;
    prow[j] = (bf16_t)p;
  }
}

// vmean[b,t,d] = cumsum_t(a)/(t+1); one thread per (b,d).
__global__ void xsa_cummean_kernel(const float* __restrict__ a,
                                   float* __restrict__ vm) {
  int idx = blockIdx.x * 256 + threadIdx.x;
  if (idx >= kB * kD) return;
  int b = idx / kD, d = idx % kD;
  float acc = 0.f;
  for (int t = 0; t < kT; ++t) {
    acc += a[((size_t)b * kT + t) * kD + d];
    vm[((size_t)b * kT + t) * kD + d] = acc / (float)(t + 1);
  }
}

__global__ void gate_combine_kernel(float* __restrict__ attn_o,
                                    const float* __restrict__ xsa_o,
                                    const float* __restrict__ gate,
                                    long long n) {
  long long i = (long long)blockIdx.x * 256 + threadIdx.x;
  if (i >= n) return;
  int d = (int)(i % kD);
  float s = 1.f / (1.f + __expf(-gate[d]));
  attn_o[i] -= s * xsa_o[i];
}

__global__ void act_kernel(float* __restrict__ h, long long n) {
  long long i = (long long)blockIdx.x * 256 + threadIdx.x;
  if (i >= n) return;
  float v = h[i];
  v = (v >= 0.f) ? v : 0.5f * v;  // leaky_relu(0.5)
  h[i] = v * v;                   // squared
}

__global__ void softcap_kernel(float* __restrict__ y, long long n) {
  long long i = (long long)blockIdx.x * 256 + threadIdx.x;
  if (i < n) y[i] = kCap * tanhf(y[i] / kCap);
}

}  // namespace

// ---------------- host orchestration ---------------------------------------

extern "C" void kernel_launch(void* const* d_in, const int* in_sizes, int n_in,
                              void* d_out, int out_size, void* d_ws, size_t ws_size,
                              hipStream_t stream) {
  (void)in_sizes; (void)n_in; (void)out_size; (void)ws_size;

  // ---- input index map (setup_inputs insertion order) ----
  auto P = [&](int i) { return (const float*)d_in[i]; };
  auto L = [&](int l, int j) { return (const float*)d_in[13 + l * 14 + j]; };
  enum { L_LN_ATTN = 0, L_LN_MLP, L_QW, L_QG, L_KW, L_KG, L_VW,
         L_OUTW, L_OUTG, L_QKGAIN, L_FCW, L_FCG, L_PROJW, L_PROJG };

  // ---- workspace layout (~132 MB) ----
  char* ws = (char*)d_ws;
  size_t off = 0;
  auto alloc = [&](size_t bytes) -> char* {
    char* p = ws + off;
    off += (bytes + 255) & ~(size_t)255;
    return p;
  };
  float*  x      = (float*)alloc((size_t)kM * kD * 4);
  float*  a_in   = (float*)alloc((size_t)kM * kD * 4);
  float*  hdn    = (float*)alloc((size_t)kM * kHID * 4);
  float*  q32    = (float*)alloc((size_t)kM * kD * 4);
  float*  k32    = (float*)alloc((size_t)kM * kKVD * 4);
  float*  v32    = (float*)alloc((size_t)kM * kKVD * 4);
  float*  o32    = (float*)alloc((size_t)kM * kD * 4);
  float*  attn_o = (float*)alloc((size_t)kM * kD * 4);
  float*  vmean  = (float*)alloc((size_t)kM * kD * 4);
  float*  xsa_o  = (float*)alloc((size_t)kM * kD * 4);
  float*  mlp_o  = (float*)alloc((size_t)kM * kD * 4);
  float*  scores = (float*)alloc((size_t)kZH * kT * kT * 4);
  bf16_t* probs  = (bf16_t*)alloc((size_t)kZH * kT * kT * 2);
  bf16_t* bufA   = (bf16_t*)alloc((size_t)kM * kHID * 2);
  bf16_t* bufB   = (bf16_t*)alloc((size_t)kV * kD * 2);
  bf16_t* qb     = (bf16_t*)alloc((size_t)kM * kD * 2);
  bf16_t* kbuf   = (bf16_t*)alloc((size_t)kM * kKVD * 2);
  bf16_t* vbuf   = (bf16_t*)alloc((size_t)kM * kKVD * 2);
  float*  bg32   = (float*)alloc((size_t)kM * kBigD * 4);
  float*  parts  = (float*)alloc(256 * 4);
  float*  thr    = (float*)alloc(4);

  auto blocks = [](long long n) { return (unsigned)((n + 255) / 256); };

  // TDM-staged GEMM: M%64==0, N%64==0, K%128==0 (true for all call sites).
  auto gemm = [&](const bf16_t* A, int lda, const bf16_t* Bm, int ldb,
                  float* C, int ldc, int M, int N, int K) {
    gemm_rr_kernel<<<dim3(N / 64, M / 64), 128, 0, stream>>>(
        A, lda, Bm, ldb, C, ldc, M, N, K);
  };

  // bitlinear: C[M,N] = rmsnorm(xin, g) @ ternary(W[N,K]).T
  auto bitlinear = [&](const float* W, const float* g, const float* xin,
                       int N, int K, int Kpad, float* C) {
    long long nw = (long long)N * K;
    absmean_partial<<<256, 256, 0, stream>>>(W, parts, nw);
    absmean_final<<<1, 256, 0, stream>>>(parts, thr, 1.0f / (float)nw);
    ternary_bf16_kernel<<<blocks((long long)N * Kpad), 256, 0, stream>>>(
        W, thr, bufB, N, K, Kpad);
    rmsnorm_bf16_kernel<<<kM, 256, 0, stream>>>(xin, g, bufA, K, Kpad);
    gemm(bufA, Kpad, bufB, Kpad, C, N, kM, N, Kpad);
  };

  // ===== embedding cell =====
  embed_kernel<<<kM, 256, 0, stream>>>((const long long*)d_in[0], P(1), P(2),
                                       x, bg32);
  bitlinear(P(3), P(4), bg32, kD, kBigD, kBigP, attn_o);        // bigram_proj
  add2_kernel<<<blocks((long long)kM * kD), 256, 0, stream>>>(
      a_in, x, attn_o, (long long)kM * kD);                     // x + bg
  bitlinear(P(5), P(6), a_in, kD, kD, kD, x);                   // embed_proj

  // ===== transformer layers =====
  for (int l = 0; l < kNL; ++l) {
    // attention cell
    rmsnorm_f32_kernel<<<kM, 256, 0, stream>>>(x, L(l, L_LN_ATTN), a_in, kD);
    bitlinear(L(l, L_QW), L(l, L_QG), a_in, kD, kD, kD, q32);
    bitlinear(L(l, L_KW), L(l, L_KG), a_in, kKVD, kD, kD, k32);
    // v projection (plain weights -> bf16)
    f32_to_bf16_kernel<<<blocks((long long)kKVD * kD), 256, 0, stream>>>(
        L(l, L_VW), bufB, (long long)kKVD * kD);
    f32_to_bf16_kernel<<<blocks((long long)kM * kD), 256, 0, stream>>>(
        a_in, bufA, (long long)kM * kD);
    gemm(bufA, kD, bufB, kD, v32, kKVD, kM, kKVD, kD);
    // quantize q/k/v for attention WMMA
    f32_to_bf16_kernel<<<blocks((long long)kM * kD), 256, 0, stream>>>(
        q32, qb, (long long)kM * kD);
    f32_to_bf16_kernel<<<blocks((long long)kM * kKVD), 256, 0, stream>>>(
        k32, kbuf, (long long)kM * kKVD);
    f32_to_bf16_kernel<<<blocks((long long)kM * kKVD), 256, 0, stream>>>(
        v32, vbuf, (long long)kM * kKVD);
    attn_scores_kernel<<<dim3(kT / 256, kT / 16, kZH), 128, 0, stream>>>(
        qb, kbuf, scores);
    softmax_kernel<<<dim3(kT, kZH), 32, 0, stream>>>(scores, probs,
                                                     L(l, L_QKGAIN));
    attn_o_kernel<<<dim3(1, kT / 16, kZH), 32, 0, stream>>>(probs, vbuf, o32);
    bitlinear(L(l, L_OUTW), L(l, L_OUTG), o32, kD, kD, kD, attn_o);
    // context (XSA) cell
    xsa_cummean_kernel<<<blocks(kB * kD), 256, 0, stream>>>(attn_o, vmean);
    bitlinear(P(7), P(8), vmean, kD, kD, kD, xsa_o);
    gate_combine_kernel<<<blocks((long long)kM * kD), 256, 0, stream>>>(
        attn_o, xsa_o, P(9), (long long)kM * kD);
    // transform (MLP) cell
    rmsnorm_f32_kernel<<<kM, 256, 0, stream>>>(x, L(l, L_LN_MLP), a_in, kD);
    bitlinear(L(l, L_FCW), L(l, L_FCG), a_in, kHID, kD, kD, hdn);
    act_kernel<<<blocks((long long)kM * kHID), 256, 0, stream>>>(
        hdn, (long long)kM * kHID);
    bitlinear(L(l, L_PROJW), L(l, L_PROJG), hdn, kD, kHID, kHID, mlp_o);
    // residual
    add3_kernel<<<blocks((long long)kM * kD), 256, 0, stream>>>(
        x, attn_o, mlp_o, (long long)kM * kD);
  }

  // ===== prediction cell =====
  rmsnorm_f32_kernel<<<kM, 256, 0, stream>>>(x, P(10), a_in, kD);
  bitlinear(P(11), P(12), a_in, kD, kD, kD, q32);               // xh
  f32_to_bf16_kernel<<<blocks((long long)kM * kD), 256, 0, stream>>>(
      q32, bufA, (long long)kM * kD);
  f32_to_bf16_kernel<<<blocks((long long)kV * kD), 256, 0, stream>>>(
      P(1), bufB, (long long)kV * kD);                          // tok_emb (tied)
  gemm(bufA, kD, bufB, kD, (float*)d_out, kV, kM, kV, kD);      // logits
  softcap_kernel<<<blocks((long long)kM * kV), 256, 0, stream>>>(
      (float*)d_out, (long long)kM * kV);
}
